// MultiHeadAttention3D_4930622456087
// MI455X (gfx1250) — matrix operand order repaired and checked by reading, not compiled, not run on hardware
//
#include <hip/hip_runtime.h>
#include <stdint.h>

#define SLEN  1024
#define BSZ   8
#define EMB   1024
#define NHEAD 16
#define HDIM  64
#define BHEAD (BSZ*NHEAD)

typedef __bf16 bf16_t;
typedef __attribute__((ext_vector_type(16))) __bf16 v16bf;
typedef __attribute__((ext_vector_type(8)))  float  v8f;

// ---------------------------------------------------------------------------
// WMMA helper: D = A(16x32 bf16) x B(32x16 bf16) + C(16x16 f32)
// ---------------------------------------------------------------------------
__device__ __forceinline__ v8f wmma_bf16(v16bf a, v16bf b, v8f c) {
  return __builtin_amdgcn_wmma_f32_16x16x32_bf16(
      /*neg_a=*/false, a, /*neg_b=*/false, b,
      /*c_mod=*/(short)0, c, /*reuse_a=*/false, /*reuse_b=*/false);
}

// Load 16 bf16 (two 16B chunks) into a fragment via b128 loads.
__device__ __forceinline__ v16bf ld_frag2(const bf16_t* p0, const bf16_t* p1) {
  union { uint4 u[2]; v16bf v; } t;
  t.u[0] = *reinterpret_cast<const uint4*>(p0);
  t.u[1] = *reinterpret_cast<const uint4*>(p1);
  return t.v;
}

// A fragment: 16x32 (MxK), row-major source with leading dim `ld` (elements).
// Per ISA 7.12.2: lane half h -> VGPR0-3 hold K=8h..8h+7, VGPR4-7 hold K=16+8h..
__device__ __forceinline__ v16bf ld_frag_a(const bf16_t* src, int ld, int k0, int lane) {
  const int half = lane >> 4, m = lane & 15;
  const bf16_t* p = src + (size_t)m * ld + k0 + 8 * half;
  return ld_frag2(p, p + 16);
}

// B fragment: Bmat(32x16, KxN) with Bmat[k][n] = src[n*ld + k0 + k]
// (src rows are the N dimension, contiguous along K).
__device__ __forceinline__ v16bf ld_frag_b(const bf16_t* src, int ld, int k0, int lane) {
  const int g = lane >> 4, n = lane & 15;
  const bf16_t* p = src + (size_t)n * ld + k0 + 8 * g;
  return ld_frag2(p, p + 16);
}

// ---------------------------------------------------------------------------
// CDNA5 async global -> LDS copy (ASYNCcnt path, per CDNA5 ISA).
// One instruction moves 16B per active lane directly into LDS.
// Generic LDS pointer's low 32 bits are the wave-relative LDS byte address.
// ---------------------------------------------------------------------------
__device__ __forceinline__ void async_copy_b128(void* lds_ptr, const void* gptr) {
  uint32_t lds = (uint32_t)(uintptr_t)lds_ptr;
  asm volatile("global_load_async_to_lds_b128 %0, %1, off"
               :: "v"(lds), "v"(gptr) : "memory");
}
__device__ __forceinline__ void wait_async0() {
  asm volatile("s_wait_asynccnt 0" ::: "memory");
}

// Immediate-pattern 16-lane butterfly via ds_swizzle (group-of-32, xor mode).
template <int XORM>
__device__ __forceinline__ float swz_xor(float x) {
  return __builtin_bit_cast(float,
      __builtin_amdgcn_ds_swizzle(__builtin_bit_cast(int, x), (XORM << 10) | 0x1f));
}

// ---------------------------------------------------------------------------
// fp32 -> bf16 bulk convert (vectorized: float4 in, 4x bf16 out)
// ---------------------------------------------------------------------------
__global__ __launch_bounds__(256)
void cvt_f32_to_bf16(const float* __restrict__ src, bf16_t* __restrict__ dst, int n4) {
  int i = blockIdx.x * blockDim.x + threadIdx.x;
  if (i < n4) {
    float4 f = reinterpret_cast<const float4*>(src)[i];
    union { bf16_t h[4]; uint2 u; } o;
    o.h[0] = (bf16_t)f.x; o.h[1] = (bf16_t)f.y;
    o.h[2] = (bf16_t)f.z; o.h[3] = (bf16_t)f.w;
    reinterpret_cast<uint2*>(dst)[i] = o.u;
  }
}

// ---------------------------------------------------------------------------
// GEMM: C[M=8192, N=1024] = A[M,1024] @ W[N,1024]^T   (bf16 in, f32 accum)
// Block = 8 waves (2x4): block tile 64(M) x 256(N); wave tile 32x64.
// K is streamed 32-deep through double-buffered LDS panels filled with
// global_load_async_to_lds_b128 (copy of step k+1 overlaps WMMAs of step k).
// grid = (M/64 = 128, N/256 = 4).
// MODE 1: store bf16 head-split  out[(b*16+h)][s][d]   (Q, K)
// MODE 2: store bf16 transposed  out[(b*16+h)][d][s]   (V)
// MODE 3: store f32 row-major    out[m][n]             (final projection)
// ---------------------------------------------------------------------------
template <int MODE>
__global__ __launch_bounds__(256)
void gemm_xWT(const bf16_t* __restrict__ A, const bf16_t* __restrict__ W,
              void* __restrict__ outp) {
  __shared__ __align__(16) bf16_t sA[2][64 * 32];    //  4 KB per buffer
  __shared__ __align__(16) bf16_t sB[2][256 * 32];   // 16 KB per buffer

  const int tid  = threadIdx.x;
  const int lane = tid & 31;
  const int wid  = tid >> 5;                 // 0..7
  const int wm = wid >> 2, wn = wid & 3;     // wave tile: rows wm*32, cols wn*64
  const int rowBase = blockIdx.x * 64;
  const int colBase = blockIdx.y * 256;

  v8f acc[2][4];
#pragma unroll
  for (int a = 0; a < 2; ++a)
#pragma unroll
    for (int b = 0; b < 4; ++b)
#pragma unroll
      for (int i = 0; i < 8; ++i) acc[a][b][i] = 0.0f;

  // 5 async b128 chunks per thread per K-step: 1 for the A panel, 4 for B.
  auto issue = [&](int buf, int k0) {
    {
      const int r = tid >> 2, o = (tid & 3) * 8;
      async_copy_b128(&sA[buf][tid * 8],
                      A + (size_t)(rowBase + r) * EMB + k0 + o);
    }
#pragma unroll
    for (int j = 0; j < 4; ++j) {
      const int cb = tid + 256 * j;
      const int r = cb >> 2, o = (cb & 3) * 8;
      async_copy_b128(&sB[buf][cb * 8],
                      W + (size_t)(colBase + r) * EMB + k0 + o);
    }
  };

  issue(0, 0);
  wait_async0();
  __syncthreads();

  for (int it = 0; it < EMB / 32; ++it) {
    const int buf = it & 1;
    if (it + 1 < EMB / 32) issue(buf ^ 1, (it + 1) * 32);

    const bf16_t* aP = &sA[buf][(wm * 32) * 32];
    const v16bf a0 = ld_frag_a(aP,            32, 0, lane);
    const v16bf a1 = ld_frag_a(aP + 16 * 32,  32, 0, lane);
    const bf16_t* bP = &sB[buf][(wn * 64) * 32];
#pragma unroll
    for (int ni = 0; ni < 4; ++ni) {
      const v16bf b = ld_frag_b(bP + ni * 16 * 32, 32, 0, lane);
      acc[0][ni] = wmma_bf16(a0, b, acc[0][ni]);
      acc[1][ni] = wmma_bf16(a1, b, acc[1][ni]);
    }
    wait_async0();       // next panel landed (covers this wave's issues)
    __syncthreads();     // all waves done reading + all panels visible
  }

  const int half = lane >> 4, nl = lane & 15;
#pragma unroll
  for (int mi = 0; mi < 2; ++mi)
#pragma unroll
    for (int ni = 0; ni < 4; ++ni)
#pragma unroll
      for (int i = 0; i < 8; ++i) {
        const int m = rowBase + wm * 32 + 16 * mi + i + 8 * half;  // m = s*BSZ+b
        const int n = colBase + wn * 64 + 16 * ni + nl;            // n = h*64+d
        const float vv = acc[mi][ni][i];
        if (MODE == 3) {
          reinterpret_cast<float*>(outp)[(size_t)m * EMB + n] = vv;
        } else {
          const int s = m >> 3, b = m & 7, h = n >> 6, d = n & 63;
          bf16_t* o = reinterpret_cast<bf16_t*>(outp);
          if (MODE == 1)
            o[(((size_t)(b * NHEAD + h)) * SLEN + s) * HDIM + d] = (bf16_t)vv;
          else  // MODE == 2: V stored transposed [bh][d][s]
            o[(((size_t)(b * NHEAD + h)) * HDIM + d) * SLEN + s] = (bf16_t)vv;
        }
      }
}

// ---------------------------------------------------------------------------
// Flash attention: one wave handles 16 query rows of one (b,h).
// Qh/Kh: [BH][S][64] bf16;  Vt: [BH][64][S] bf16;  mask: [BH][S][S] bytes.
// attnC: [S*B][E] bf16 (input to the final projection).
// block = 4 waves; grid = (BH, S/64).
// ---------------------------------------------------------------------------
__global__ __launch_bounds__(128)
void attn_flash(const bf16_t* __restrict__ Qh, const bf16_t* __restrict__ Kh,
                const bf16_t* __restrict__ Vt, const uint8_t* __restrict__ mask,
                bf16_t* __restrict__ attnC) {
  __shared__ __align__(16) bf16_t Pbuf[4][16][32];

  const int lane = threadIdx.x & 31;
  const int wid  = threadIdx.x >> 5;
  const int bh    = blockIdx.x;
  const int qbase = (blockIdx.y * 4 + wid) * 16;
  const int half = lane >> 4, nl = lane & 15;
  const float scale = 0.125f;  // 1/sqrt(64)

  // Q tile stays resident: two K=32 A-fragments covering d=0..63
  const bf16_t* Qrow = Qh + ((size_t)bh * SLEN + qbase) * HDIM;
  const v16bf qa0 = ld_frag_a(Qrow, HDIM, 0,  lane);
  const v16bf qa1 = ld_frag_a(Qrow, HDIM, 32, lane);

  v8f o[4];
  float mrow[8], lrow[8];
#pragma unroll
  for (int i = 0; i < 8; ++i) { mrow[i] = -1e30f; lrow[i] = 0.0f; }
#pragma unroll
  for (int nt = 0; nt < 4; ++nt)
#pragma unroll
    for (int i = 0; i < 8; ++i) o[nt][i] = 0.0f;

  const uint8_t* mbase = mask + ((size_t)bh * SLEN + qbase) * SLEN;

  for (int jb = 0; jb < SLEN; jb += 32) {
    float sv[2][8];
    bool  mf[2][8];
    // ---- S = Q K^T for two 16-column tiles ----
#pragma unroll
    for (int t = 0; t < 2; ++t) {
      const int jt = jb + 16 * t;
      const bf16_t* Krow = Kh + ((size_t)bh * SLEN + jt) * HDIM;
      v8f s;
#pragma unroll
      for (int i = 0; i < 8; ++i) s[i] = 0.0f;
      s = wmma_bf16(qa0, ld_frag_b(Krow, HDIM, 0,  lane), s);
      s = wmma_bf16(qa1, ld_frag_b(Krow, HDIM, 32, lane), s);
#pragma unroll
      for (int i = 0; i < 8; ++i) {
        const int ri = i + 8 * half;
        const uint8_t mk = mbase[(size_t)ri * SLEN + jt + nl];
        mf[t][i] = (mk != 0);
        sv[t][i] = mf[t][i] ? -1e30f : s[i] * scale;
      }
    }
    // ---- online softmax: row max across the 16 lanes of each half ----
    float f[8];
#pragma unroll
    for (int i = 0; i < 8; ++i) {
      float tm = fmaxf(sv[0][i], sv[1][i]);
      tm = fmaxf(tm, swz_xor<1>(tm));
      tm = fmaxf(tm, swz_xor<2>(tm));
      tm = fmaxf(tm, swz_xor<4>(tm));
      tm = fmaxf(tm, swz_xor<8>(tm));
      const float mnew = fmaxf(mrow[i], tm);
      f[i] = __expf(mrow[i] - mnew);
      mrow[i] = mnew;
    }
    // ---- P = exp(S - m), masked -> 0; stage through LDS for re-layout ----
#pragma unroll
    for (int t = 0; t < 2; ++t)
#pragma unroll
      for (int i = 0; i < 8; ++i) {
        const int ri = i + 8 * half;
        const float p = mf[t][i] ? 0.0f : __expf(sv[t][i] - mrow[i]);
        sv[t][i] = p;
        Pbuf[wid][ri][16 * t + nl] = (bf16_t)p;
      }
#pragma unroll
    for (int i = 0; i < 8; ++i) {
      float rs = sv[0][i] + sv[1][i];
      rs += swz_xor<1>(rs);
      rs += swz_xor<2>(rs);
      rs += swz_xor<4>(rs);
      rs += swz_xor<8>(rs);
      lrow[i] = lrow[i] * f[i] + rs;
#pragma unroll
      for (int nt = 0; nt < 4; ++nt) o[nt][i] *= f[i];
    }
    __syncthreads();
    // ---- O += P (16x32) @ V (32x64): A-frag from LDS, B-frags from Vt ----
    const v16bf pa = ld_frag_a(&Pbuf[wid][0][0], 32, 0, lane);
#pragma unroll
    for (int nt = 0; nt < 4; ++nt) {
      const bf16_t* Vrow = Vt + ((size_t)bh * HDIM + nt * 16) * SLEN;
      o[nt] = wmma_bf16(pa, ld_frag_b(Vrow, SLEN, jb, lane), o[nt]);
    }
    __syncthreads();
  }

  // ---- normalize and scatter to [s*BSZ+b][h*64+d] bf16 ----
  const int b = bh >> 4, h = bh & 15;
#pragma unroll
  for (int i = 0; i < 8; ++i) {
    const float inv = (lrow[i] > 0.0f) ? __builtin_amdgcn_rcpf(lrow[i]) : 0.0f;
    const int s = qbase + i + 8 * half;
#pragma unroll
    for (int nt = 0; nt < 4; ++nt) {
      const int e = h * HDIM + nt * 16 + nl;
      attnC[((size_t)s * BSZ + b) * EMB + e] = (bf16_t)(o[nt][i] * inv);
    }
  }
}

// ---------------------------------------------------------------------------
// Host orchestration.
// Inputs: 0=q 1=k 2=v 3=mask(bool bytes) 4=Wq 5=Wk 6=Wv 7=Wo
// Workspace (104 MB): bf16 copies of activations/weights + head-split Q/K/Vt.
// attnC reuses the qb slot (qb dead after projections).
// ---------------------------------------------------------------------------
extern "C" void kernel_launch(void* const* d_in, const int* in_sizes, int n_in,
                              void* d_out, int out_size, void* d_ws, size_t ws_size,
                              hipStream_t stream) {
  (void)in_sizes; (void)n_in; (void)out_size; (void)ws_size;
  const float*   q    = (const float*)d_in[0];
  const float*   k    = (const float*)d_in[1];
  const float*   v    = (const float*)d_in[2];
  const uint8_t* mask = (const uint8_t*)d_in[3];
  const float*   Wq   = (const float*)d_in[4];
  const float*   Wk   = (const float*)d_in[5];
  const float*   Wv   = (const float*)d_in[6];
  const float*   Wo   = (const float*)d_in[7];

  char* ws = (char*)d_ws;
  const size_t MB = 1024ull * 1024ull;
  bf16_t* qb    = (bf16_t*)(ws + 0 * MB);    // 16 MB, reused as attnC
  bf16_t* kb    = (bf16_t*)(ws + 16 * MB);   // 16 MB
  bf16_t* vb    = (bf16_t*)(ws + 32 * MB);   // 16 MB
  bf16_t* Wqb   = (bf16_t*)(ws + 48 * MB);   // 2 MB
  bf16_t* Wkb   = (bf16_t*)(ws + 50 * MB);
  bf16_t* Wvb   = (bf16_t*)(ws + 52 * MB);
  bf16_t* Wob   = (bf16_t*)(ws + 54 * MB);
  bf16_t* Qhp   = (bf16_t*)(ws + 56 * MB);   // [128][1024][64]
  bf16_t* Khp   = (bf16_t*)(ws + 72 * MB);   // [128][1024][64]
  bf16_t* Vtp   = (bf16_t*)(ws + 88 * MB);   // [128][64][1024]
  bf16_t* attnC = qb;                        // [8192][1024]

  auto cvt = [&](const float* s, bf16_t* d, int n) {
    const int n4 = n / 4;
    cvt_f32_to_bf16<<<(n4 + 255) / 256, 256, 0, stream>>>(s, d, n4);
  };
  cvt(q,  qb,  SLEN * BSZ * EMB);
  cvt(k,  kb,  SLEN * BSZ * EMB);
  cvt(v,  vb,  SLEN * BSZ * EMB);
  cvt(Wq, Wqb, EMB * EMB);
  cvt(Wk, Wkb, EMB * EMB);
  cvt(Wv, Wvb, EMB * EMB);
  cvt(Wo, Wob, EMB * EMB);

  const dim3 gg(128, 4), gb(256);           // M/64 = 128, N/256 = 4
  gemm_xWT<1><<<gg, gb, 0, stream>>>(qb, Wqb, Qhp);
  gemm_xWT<1><<<gg, gb, 0, stream>>>(kb, Wkb, Khp);
  gemm_xWT<2><<<gg, gb, 0, stream>>>(vb, Wvb, Vtp);

  attn_flash<<<dim3(BHEAD, SLEN / 64), 128, 0, stream>>>(Qhp, Khp, Vtp, mask, attnC);

  gemm_xWT<3><<<gg, gb, 0, stream>>>(attnC, Wob, d_out);
}
